// ModelDSGPS_79370995631014
// MI455X (gfx1250) — compile-verified, matrix operand order checked by reading, not run on hardware
//
#include <hip/hip_runtime.h>
#include <hip/hip_bf16.h>

// ---------------------------------------------------------------------------
// DS-GPS GNN forward for MI455X (gfx1250, wave32).
// Edge message MLPs (131->64->64, x2 phis, x2 steps, 800K edges) dominate
// (~80 GFLOP) and are lowered to v_wmma_f32_16x16x32_f16 with f32 accumulate.
// Weights are pre-packed into WMMA B-fragment order (f16); edge feature tiles
// (16 edges x 160 padded K) are staged in LDS as f16 via vectorized b128
// gathers and read as A fragments per the CDNA5 16-bit A layout
// (lane%16 = M row, lane/16 selects K half, two 8-half runs -> 2x ds_load_b128).
// ---------------------------------------------------------------------------

typedef __attribute__((ext_vector_type(16))) _Float16 v16h;
typedef __attribute__((ext_vector_type(8)))  _Float16 v8h;
typedef __attribute__((ext_vector_type(4)))  _Float16 v4h;
typedef __attribute__((ext_vector_type(8)))  float    v8f;
typedef __attribute__((ext_vector_type(4)))  float    v4f;

#define LAT  64
#define KPAD 160   // 2*64+3 = 131 padded to 5 chunks of 32

// ---------- weight repack into WMMA B-fragment order (f16) ------------------
// Consumption: frag[((kc*4+nt)*32 + lane)*16 + e] where for lane l, element e:
//   n = nt*16 + (l&15),  k = kc*32 + (e<8 ? (l>>4)*8+e : 16+(l>>4)*8+(e-8))
__global__ void pack_frag_kernel(const float* __restrict__ W,
                                 _Float16* __restrict__ out,
                                 int K, int kchunks) {
    int idx = blockIdx.x * blockDim.x + threadIdx.x;
    int total = kchunks * 4 * 32 * 16;
    if (idx >= total) return;
    int e    = idx & 15;
    int lane = (idx >> 4) & 31;
    int t    = idx >> 9;
    int nt   = t & 3;
    int kc   = t >> 2;
    int g    = lane >> 4;
    int n    = (nt << 4) + (lane & 15);
    int k    = kc * 32 + ((e < 8) ? (g * 8 + e) : (16 + g * 8 + (e - 8)));
    float v  = (k < K) ? W[k * LAT + n] : 0.0f;
    out[idx] = (_Float16)v;
}

// ---------- encoder: H0 = relu(x*eW1+eb1) @ eW2 + eb2 -----------------------
__global__ __launch_bounds__(64) void encoder_kernel(
    const float* __restrict__ x,
    const float* __restrict__ eW1, const float* __restrict__ eb1,
    const float* __restrict__ eW2, const float* __restrict__ eb2,
    float* __restrict__ H, float* __restrict__ H0) {
    const int node = blockIdx.x;
    const int t = threadIdx.x;
    __shared__ float hid[LAT];
    float xv = x[node];
    hid[t] = fmaxf(xv * eW1[t] + eb1[t], 0.0f);
    __syncthreads();
    float s = eb2[t];
    for (int k = 0; k < LAT; ++k) s += hid[k] * eW2[k * LAT + t];
    H[(size_t)node * LAT + t] = s;
    H0[(size_t)node * LAT + t] = s;
}

// ---------- one phi (message MLP) on a 16-edge tile, WMMA -------------------
__device__ __forceinline__ void run_phi(
    const _Float16* __restrict__ arow,   // LDS: &Atile[w][m16][0]
    _Float16* __restrict__ hbase,        // LDS: &Htile[w][0][0] (16x64)
    int m16, int g, int lane,
    const _Float16* __restrict__ w1frag, const float* __restrict__ b1,
    const _Float16* __restrict__ w2frag, const float* __restrict__ b2,
    const float* __restrict__ maskp, const int* __restrict__ idxp,
    float* __restrict__ Mout, bool swapK) {
    // GEMM1: A[16x160] x W1[160x64] -> hidden, +bias, ReLU, f16 -> LDS
#pragma unroll
    for (int nt = 0; nt < 4; ++nt) {
        v8f acc = {};
#pragma unroll
        for (int kc = 0; kc < 5; ++kc) {
            int k0 = kc * 32 + g * 8;
            int k1 = k0 + 16;
            // phi_from = same tile with the two H 64-blocks swapped (k^64)
            int k0s = (swapK && k0 < 128) ? (k0 ^ 64) : k0;
            int k1s = (swapK && k1 < 128) ? (k1 ^ 64) : k1;
            v8h alo = *(const v8h*)(arow + k0s);
            v8h ahi = *(const v8h*)(arow + k1s);
            v16h a;
#pragma unroll
            for (int i = 0; i < 8; ++i) { a[i] = alo[i]; a[i + 8] = ahi[i]; }
            v16h b = *(const v16h*)(w1frag + ((kc * 4 + nt) * 32 + lane) * 16);
            acc = __builtin_amdgcn_wmma_f32_16x16x32_f16(
                false, a, false, b, (short)0, acc, false, false);
        }
#pragma unroll
        for (int v = 0; v < 8; ++v) {
            int mm = (g << 3) + v;                 // C/D layout: M = v + 8*(lane/16)
            float hv = acc[v] + b1[(nt << 4) + m16];
            hv = fmaxf(hv, 0.0f);
            hbase[mm * LAT + (nt << 4) + m16] = (_Float16)hv;
        }
    }
    __syncthreads();
    // GEMM2: hidden[16x64] x W2[64x64] -> out, +bias, mask, atomic scatter
    const _Float16* hrow = hbase + m16 * LAT;
#pragma unroll
    for (int nt = 0; nt < 4; ++nt) {
        v8f acc = {};
#pragma unroll
        for (int kc = 0; kc < 2; ++kc) {
            int k0 = kc * 32 + g * 8;
            v8h alo = *(const v8h*)(hrow + k0);
            v8h ahi = *(const v8h*)(hrow + k0 + 16);
            v16h a;
#pragma unroll
            for (int i = 0; i < 8; ++i) { a[i] = alo[i]; a[i + 8] = ahi[i]; }
            v16h b = *(const v16h*)(w2frag + ((kc * 4 + nt) * 32 + lane) * 16);
            acc = __builtin_amdgcn_wmma_f32_16x16x32_f16(
                false, a, false, b, (short)0, acc, false, false);
        }
#pragma unroll
        for (int v = 0; v < 8; ++v) {
            int mm = (g << 3) + v;
            float o = (acc[v] + b2[(nt << 4) + m16]) * maskp[mm];
            atomicAdd(&Mout[(size_t)idxp[mm] * LAT + (nt << 4) + m16], o);
        }
    }
    __syncthreads();
}

// ---------- fused edge-message kernel: both phis share one gathered tile ----
__global__ __launch_bounds__(128) void edge_msg_kernel(
    const float* __restrict__ H,
    const int* __restrict__ rowp, const int* __restrict__ colp,
    const float* __restrict__ eattr,
    const _Float16* __restrict__ w1t, const float* __restrict__ b1t,
    const _Float16* __restrict__ w2t, const float* __restrict__ b2t,
    const _Float16* __restrict__ w1f, const float* __restrict__ b1f,
    const _Float16* __restrict__ w2f, const float* __restrict__ b2f,
    float* __restrict__ Mt, float* __restrict__ Mf, int nE) {
    __shared__ alignas(16) _Float16 Atile[4][16][KPAD];  // 20.0 KB
    __shared__ alignas(16) _Float16 Htile[4][16][LAT];   //  8.0 KB
    __shared__ int   sIto[4][16];
    __shared__ int   sIfr[4][16];
    __shared__ float sMask[4][16];

    const int w    = threadIdx.x >> 5;
    const int lane = threadIdx.x & 31;
    const int m16  = lane & 15;
    const int g    = lane >> 4;
    const int tile = blockIdx.x * 4 + w;
    const int e0   = tile * 16;

    // --- preload edge indices (one lane per edge) + attr/pad tail ----------
    if (lane < 16) {
        int e = e0 + lane;
        bool valid = e < nE;
        int ee = valid ? e : 0;
        int ci = colp[ee];
        int ri = rowp[ee];
        sIto[w][lane]  = ci;   // phi_to aggregates at i = col
        sIfr[w][lane]  = ri;   // phi_from aggregates at i = row
        sMask[w][lane] = (valid && (ci != ri)) ? 1.0f : 0.0f;  // self-loop mask
        // tail of this edge's A row: [attr0, attr1, attr2, 0...0] (halves 128..159)
        float a0 = eattr[(size_t)ee * 3 + 0];
        float a1 = eattr[(size_t)ee * 3 + 1];
        float a2 = eattr[(size_t)ee * 3 + 2];
        _Float16* dst = &Atile[w][lane][128];
        v4h t0; t0[0] = (_Float16)a0; t0[1] = (_Float16)a1;
        t0[2] = (_Float16)a2; t0[3] = (_Float16)0.0f;
        *(v4h*)(dst) = t0;                 // halves 128..131
        *(v4h*)(dst + 4) = (v4h)(_Float16)0.0f;   // 132..135
        v8h z = {};
        *(v8h*)(dst + 8)  = z;             // 136..143
        *(v8h*)(dst + 16) = z;             // 144..151
        *(v8h*)(dst + 24) = z;             // 152..159
    }
    __syncthreads();

    // --- vectorized H gather: lane = (half, quarter); one b128 per row-quarter
    // lanes 0..15 cover H[col] (features 0..63), lanes 16..31 H[row] (64..127)
    {
        const int q = lane & 15;           // float4 index within the 64-wide row
        const int fbase = (lane & 16) << 2;   // 0 or 64
        for (int m = 0; m < 16; ++m) {
            int src = (lane < 16) ? sIto[w][m] : sIfr[w][m];
            v4f v = *(const v4f*)(H + (size_t)src * LAT + q * 4);
            v4h hv;
#pragma unroll
            for (int i = 0; i < 4; ++i) hv[i] = (_Float16)v[i];
            *(v4h*)(&Atile[w][m][fbase + q * 4]) = hv;
        }
    }
    __syncthreads();

    const _Float16* arow = &Atile[w][m16][0];
    _Float16* hbase = &Htile[w][0][0];
    run_phi(arow, hbase, m16, g, lane, w1t, b1t, w2t, b2t, sMask[w], sIto[w], Mt, false);
    run_phi(arow, hbase, m16, g, lane, w1f, b1f, w2f, b2f, sMask[w], sIfr[w], Mf, true);
}

// ---------- node update: gates, Hn, U, enc/autoenc loss partials ------------
__global__ __launch_bounds__(64) void node_update_kernel(
    float* __restrict__ H, const float* __restrict__ H0,
    const float* __restrict__ Mt, const float* __restrict__ Mf,
    const float* __restrict__ prb, const int* __restrict__ tags,
    const float* __restrict__ zkW, const float* __restrict__ zkb,
    const float* __restrict__ rkW, const float* __restrict__ rkb,
    const float* __restrict__ cW,  const float* __restrict__ cb,
    const float* __restrict__ dW1, const float* __restrict__ db1,
    const float* __restrict__ dW2, const float* __restrict__ db2,
    const float* __restrict__ eW1, const float* __restrict__ eb1,
    const float* __restrict__ eW2, const float* __restrict__ eb2,
    float* __restrict__ U, float* __restrict__ acc,
    float invNL, float invN) {
    const int node = blockIdx.x;
    const int t = threadIdx.x;
    __shared__ float cat[194];
    __shared__ float shv[LAT];
    __shared__ float red[LAT];
    __shared__ float ehid[LAT];
    __shared__ float sz[LAT];
    __shared__ float su, sw;

    float hval = H[(size_t)node * LAT + t];
    cat[t]        = hval;
    cat[64 + t]   = Mt[(size_t)node * LAT + t];
    cat[128 + t]  = Mf[(size_t)node * LAT + t];
    if (t < 2) cat[192 + t] = prb[(size_t)node * 2 + t];
    __syncthreads();

    float az = zkb[t], ar = rkb[t];
    for (int k = 0; k < 194; ++k) {
        float c = cat[k];
        az += c * zkW[k * LAT + t];
        ar += c * rkW[k * LAT + t];
    }
    float alpha = 1.0f / (1.0f + __expf(-az));
    float reset = 1.0f / (1.0f + __expf(-ar));
    shv[t] = reset * hval;
    __syncthreads();
    float ac = cb[t];
    for (int k = 0; k < 64; ++k)   ac += shv[k] * cW[k * LAT + t];
    for (int k = 64; k < 194; ++k) ac += cat[k] * cW[k * LAT + t];
    float corr = tanhf(ac);
    float hn = hval + alpha * corr;
    if (tags[node] == 1) hn = H0[(size_t)node * LAT + t];   // Dirichlet reset
    H[(size_t)node * LAT + t] = hn;                          // in-place H update
    __syncthreads();
    shv[t] = hn;
    __syncthreads();
    // U = dec(Hn): relu(Hn@dW1+db1)@dW2+db2
    float dh = db1[t];
    for (int k = 0; k < 64; ++k) dh += shv[k] * dW1[k * LAT + t];
    dh = fmaxf(dh, 0.0f);
    red[t] = dh * dW2[t];
    __syncthreads();
    for (int s = 32; s > 0; s >>= 1) { if (t < s) red[t] += red[t + s]; __syncthreads(); }
    if (t == 0) { su = red[0] + db2[0]; U[node] = su; }
    __syncthreads();
    float u = su;
    // Z = enc(u)  (enc_loss = mse(enc(dec(Hn)), Hn) and dec(Hn)==U forward)
    ehid[t] = fmaxf(u * eW1[t] + eb1[t], 0.0f);
    __syncthreads();
    float z = eb2[t];
    for (int k = 0; k < 64; ++k) z += ehid[k] * eW2[k * LAT + t];
    float ed = z - hn;
    sz[t] = z;
    __syncthreads();
    // wdec = dec(Z)  (autoenc_loss = mse(dec(enc(U)), U))
    float dh2 = db1[t];
    for (int k = 0; k < 64; ++k) dh2 += sz[k] * dW1[k * LAT + t];
    dh2 = fmaxf(dh2, 0.0f);
    red[t] = dh2 * dW2[t];
    __syncthreads();
    for (int s = 32; s > 0; s >>= 1) { if (t < s) red[t] += red[t + s]; __syncthreads(); }
    if (t == 0) sw = red[0] + db2[0];
    __syncthreads();
    red[t] = ed * ed;
    __syncthreads();
    for (int s = 32; s > 0; s >>= 1) { if (t < s) red[t] += red[t + s]; __syncthreads(); }
    if (t == 0) {
        float ad = sw - u;
        atomicAdd(acc, red[0] * invNL + ad * ad * invN);
    }
}

// ---------- residual: Au[row] += a_ij * U[col]; then mean((Au-y)^2) ---------
__global__ __launch_bounds__(256) void residual_edge_kernel(
    const int* __restrict__ rowp, const int* __restrict__ colp,
    const float* __restrict__ a, const float* __restrict__ U,
    float* __restrict__ Au, int nE) {
    int e = blockIdx.x * blockDim.x + threadIdx.x;
    if (e < nE) atomicAdd(&Au[rowp[e]], a[e] * U[colp[e]]);
}

__global__ __launch_bounds__(256) void residual_reduce_kernel(
    const float* __restrict__ Au, const float* __restrict__ y,
    float* __restrict__ acc, float scale, int n) {
    __shared__ float red[256];
    float s = 0.0f;
    for (int i = blockIdx.x * blockDim.x + threadIdx.x; i < n;
         i += gridDim.x * blockDim.x) {
        float d = Au[i] - y[i];
        s += d * d;
    }
    red[threadIdx.x] = s;
    __syncthreads();
    for (int st = 128; st > 0; st >>= 1) {
        if (threadIdx.x < st) red[threadIdx.x] += red[threadIdx.x + st];
        __syncthreads();
    }
    if (threadIdx.x == 0) atomicAdd(acc, red[0] * scale);
}

__global__ void finalize_kernel(const float* __restrict__ acc, float* __restrict__ out) {
    out[0] = acc[0];
}

// ---------------------------------------------------------------------------
extern "C" void kernel_launch(void* const* d_in, const int* in_sizes, int n_in,
                              void* d_out, int out_size, void* d_ws, size_t ws_size,
                              hipStream_t stream) {
    const float* x     = (const float*)d_in[0];
    /* d_in[1] = sol: unused by the reference */
    const float* y     = (const float*)d_in[2];
    const int*   tags  = (const int*)d_in[3];
    const int*   eidx  = (const int*)d_in[4];
    const float* eattr = (const float*)d_in[5];
    const float* a_ij  = (const float*)d_in[6];
    const float* prb   = (const float*)d_in[7];
    const float* ptW1 = (const float*)d_in[8],  *ptb1 = (const float*)d_in[9];
    const float* ptW2 = (const float*)d_in[10], *ptb2 = (const float*)d_in[11];
    const float* pfW1 = (const float*)d_in[12], *pfb1 = (const float*)d_in[13];
    const float* pfW2 = (const float*)d_in[14], *pfb2 = (const float*)d_in[15];
    const float* zkW  = (const float*)d_in[16], *zkb  = (const float*)d_in[17];
    const float* rkW  = (const float*)d_in[18], *rkb  = (const float*)d_in[19];
    const float* cW   = (const float*)d_in[20], *cb   = (const float*)d_in[21];
    const float* eW1  = (const float*)d_in[22], *eb1  = (const float*)d_in[23];
    const float* eW2  = (const float*)d_in[24], *eb2  = (const float*)d_in[25];
    const float* dW1  = (const float*)d_in[26], *db1  = (const float*)d_in[27];
    const float* dW2  = (const float*)d_in[28], *db2  = (const float*)d_in[29];

    const int N = in_sizes[0];            // 50000 nodes
    const int E = in_sizes[6];            // 800000 edges
    const int* rowp = eidx;
    const int* colp = eidx + E;

    // workspace carve-out (~52 MB)
    char* p = (char*)d_ws;
    auto take = [&](size_t bytes) -> char* {
        char* r = p;
        p += (bytes + 255) & ~(size_t)255;
        return r;
    };
    float* H   = (float*)take((size_t)N * LAT * sizeof(float));
    float* H0  = (float*)take((size_t)N * LAT * sizeof(float));
    float* Mt  = (float*)take((size_t)N * LAT * sizeof(float));
    float* Mf  = (float*)take((size_t)N * LAT * sizeof(float));
    float* Au  = (float*)take((size_t)N * sizeof(float));
    float* acc = (float*)take(256);
    _Float16* w1t = (_Float16*)take(5 * 2048 * sizeof(_Float16));
    _Float16* w2t = (_Float16*)take(2 * 2048 * sizeof(_Float16));
    _Float16* w1f = (_Float16*)take(5 * 2048 * sizeof(_Float16));
    _Float16* w2f = (_Float16*)take(2 * 2048 * sizeof(_Float16));

    float* U = (float*)d_out;        // d_out = [U (N floats), total (1 float)]
    float* totalOut = U + N;

    // pack weights into WMMA fragment order (f16)
    pack_frag_kernel<<<(5 * 2048 + 255) / 256, 256, 0, stream>>>(ptW1, w1t, 2 * LAT + 3, 5);
    pack_frag_kernel<<<(2 * 2048 + 255) / 256, 256, 0, stream>>>(ptW2, w2t, LAT, 2);
    pack_frag_kernel<<<(5 * 2048 + 255) / 256, 256, 0, stream>>>(pfW1, w1f, 2 * LAT + 3, 5);
    pack_frag_kernel<<<(2 * 2048 + 255) / 256, 256, 0, stream>>>(pfW2, w2f, LAT, 2);

    encoder_kernel<<<N, 64, 0, stream>>>(x, eW1, eb1, eW2, eb2, H, H0);
    hipMemsetAsync(acc, 0, sizeof(float), stream);

    const int nTiles  = (E + 15) / 16;
    const int eblocks = (nTiles + 3) / 4;
    const float invNL = 1.0f / ((float)N * (float)LAT);
    const float invN  = 1.0f / (float)N;
    const float gam[2] = {0.9f, 1.0f};   // GAMMA^(K-1-step), K=2

    for (int step = 0; step < 2; ++step) {
        hipMemsetAsync(Mt, 0, (size_t)N * LAT * sizeof(float), stream);
        hipMemsetAsync(Mf, 0, (size_t)N * LAT * sizeof(float), stream);
        edge_msg_kernel<<<eblocks, 128, 0, stream>>>(
            H, rowp, colp, eattr,
            w1t, ptb1, w2t, ptb2, w1f, pfb1, w2f, pfb2, Mt, Mf, E);
        node_update_kernel<<<N, 64, 0, stream>>>(
            H, H0, Mt, Mf, prb, tags,
            zkW, zkb, rkW, rkb, cW, cb, dW1, db1, dW2, db2,
            eW1, eb1, eW2, eb2, U, acc, invNL, invN);
        hipMemsetAsync(Au, 0, (size_t)N * sizeof(float), stream);
        residual_edge_kernel<<<(E + 255) / 256, 256, 0, stream>>>(rowp, colp, a_ij, U, Au, E);
        residual_reduce_kernel<<<256, 256, 0, stream>>>(Au, y, acc, gam[step] * invN, N);
    }
    finalize_kernel<<<1, 1, 0, stream>>>(acc, totalOut);
}
